// Prototype_Block_29128468201513
// MI455X (gfx1250) — matrix operand order repaired and checked by reading, not compiled, not run on hardware
//
#include <hip/hip_runtime.h>
#include <math.h>

// ---------------------------------------------------------------------------
// MI455X / gfx1250. All large GEMMs on v_wmma_f32_16x16x32_bf16 (wave32,
// f32 accum), with double-buffered LDS staging fed by the gfx1250 async
// global->LDS path (ASYNCcnt) when available.
// Roofline: ~786 GFLOP vs ~600MB HBM -> strongly compute bound -> bf16 WMMA.
// ---------------------------------------------------------------------------

typedef __bf16 bf16;
typedef __attribute__((ext_vector_type(16))) __bf16 v16bf;
typedef __attribute__((ext_vector_type(8)))  __bf16 v8bf;
typedef __attribute__((ext_vector_type(8)))  float  v8f;
typedef __attribute__((ext_vector_type(4)))  int    v4i;

#define BQ   16
#define NSEQ 4096
#define DIM  768
#define NH   12
#define HD   64
#define TT   64          // P(32) + centers(32)
#define MROWS (BQ*NSEQ)  // 65536
#define LDP  40          // padded LDS row (32 k-elems + 8 pad) -> conflict-free

#if defined(__AMDGCN__) && __has_builtin(__builtin_amdgcn_global_load_async_to_lds_b128) && __has_builtin(__builtin_amdgcn_s_wait_asynccnt)
#define USE_ASYNC_LDS 1
#else
#define USE_ASYNC_LDS 0
#endif

__device__ __forceinline__ float gelu_f(float x) {
  return 0.5f * x * (1.0f + erff(x * 0.70710678118654752f));
}
__device__ __forceinline__ float silu_f(float x) {
  return x / (1.0f + __expf(-x));
}

__device__ __forceinline__ v8f wmma_bf16(v16bf a, v16bf b, v8f c) {
  return __builtin_amdgcn_wmma_f32_16x16x32_bf16(
      /*neg_a=*/false, a, /*neg_b=*/false, b,
      /*c_mod=*/(short)0, c, /*reuse_a=*/false, /*reuse_b=*/false);
}

// 16-byte global -> LDS copy; async (ASYNCcnt) when the builtin exists.
// Probe-learned signature: (int4 addrspace(1)*, int4 addrspace(3)*, imm, imm).
__device__ __forceinline__ void copy16_g2l(const bf16* gp, bf16* lp) {
#if USE_ASYNC_LDS
  __builtin_amdgcn_global_load_async_to_lds_b128(
      (__attribute__((address_space(1))) v4i*)gp,
      (__attribute__((address_space(3))) v4i*)lp, 0, 0);
#else
  *(uint4*)lp = *(const uint4*)gp;
#endif
}
template <int N>
__device__ __forceinline__ void wait_async() {
#if USE_ASYNC_LDS
  __builtin_amdgcn_s_wait_asynccnt(N);
#endif
}

// A fragment (16x32 bf16). ISA 7.12.2: lane L (m=L&15, g=L>>4) holds
// K=[g*8,g*8+8) in VGPR0..3 and K=[16+g*8,...) in VGPR4..7.
__device__ __forceinline__ v16bf load_a_frag(const bf16* base, int ld) {
  const int lane = threadIdx.x & 31;
  const int m = lane & 15, g = lane >> 4;
  const bf16* p = base + (long long)m * ld + g * 8;
  union { v16bf v; v8bf h[2]; } f;
  f.h[0] = *(const v8bf*)(p);
  f.h[1] = *(const v8bf*)(p + 16);
  return f.v;
}

// B fragment (32x16 bf16) for C = A*W^T with W row-major [Ncols,K]:
// lane L (n=L&15, g=L>>4) holds K=[g*8,g*8+8) and K=[16+g*8,16+g*8+8).
__device__ __forceinline__ v16bf load_b_frag(const bf16* base, int ld) {
  const int lane = threadIdx.x & 31;
  const int n = lane & 15, g = lane >> 4;
  const bf16* p = base + (long long)n * ld + g * 8;
  union { v16bf v; v8bf h[2]; } f;
  f.h[0] = *(const v8bf*)(p);
  f.h[1] = *(const v8bf*)(p + 16);
  return f.v;
}

// ---------------------------------------------------------------------------
__global__ void cvt_f32_bf16(const float* __restrict__ s, bf16* __restrict__ d, int n) {
  int i = blockIdx.x * 256 + threadIdx.x;
  if (i < n) d[i] = (bf16)s[i];
}

// ---------------------------------------------------------------------------
// LayerNorm over 768 cols, one block per row.
// ---------------------------------------------------------------------------
template <typename T>
__global__ __launch_bounds__(256) void ln_kernel(
    const T* __restrict__ x, const float* __restrict__ w,
    const float* __restrict__ b, bf16* __restrict__ out) {
  __shared__ float red[256];
  const int tid = threadIdx.x;
  const T* xr = x + (long long)blockIdx.x * DIM;
  float v0 = (float)xr[tid], v1 = (float)xr[tid + 256], v2 = (float)xr[tid + 512];
  red[tid] = v0 + v1 + v2;
  __syncthreads();
  for (int o = 128; o > 0; o >>= 1) { if (tid < o) red[tid] += red[tid + o]; __syncthreads(); }
  float mean = red[0] * (1.0f / DIM);
  __syncthreads();
  float d0 = v0 - mean, d1 = v1 - mean, d2 = v2 - mean;
  red[tid] = d0 * d0 + d1 * d1 + d2 * d2;
  __syncthreads();
  for (int o = 128; o > 0; o >>= 1) { if (tid < o) red[tid] += red[tid + o]; __syncthreads(); }
  float rstd = rsqrtf(red[0] * (1.0f / DIM) + 1e-5f);
  bf16* orow = out + (long long)blockIdx.x * DIM;
  orow[tid]       = (bf16)(d0 * rstd * w[tid]       + b[tid]);
  orow[tid + 256] = (bf16)(d1 * rstd * w[tid + 256] + b[tid + 256]);
  orow[tid + 512] = (bf16)(d2 * rstd * w[tid + 512] + b[tid + 512]);
}

// ---------------------------------------------------------------------------
// Prototype enhancer MLP (512 rows, tiny).
// ---------------------------------------------------------------------------
__global__ __launch_bounds__(256) void proto_enhance(
    const bf16* __restrict__ pn, const bf16* __restrict__ We1,
    const float* __restrict__ be1, const bf16* __restrict__ We2,
    const float* __restrict__ be2, bf16* __restrict__ combined) {
  __shared__ float s_p[DIM];
  __shared__ float s_h[2 * DIM];
  const int tid = threadIdx.x;
  const int row = blockIdx.x;            // b*32 + p
  const int b = row >> 5, p = row & 31;
  const bf16* pr = pn + (long long)row * DIM;
  for (int i = tid; i < DIM; i += 256) s_p[i] = (float)pr[i];
  __syncthreads();
  for (int i = 0; i < 6; ++i) {
    int j = tid + 256 * i;
    const bf16* wr = We1 + (long long)j * DIM;
    float acc = be1[j];
    for (int k = 0; k < DIM; ++k) acc += s_p[k] * (float)wr[k];
    s_h[j] = gelu_f(acc);
  }
  __syncthreads();
  bf16* out = combined + ((long long)b * TT + p) * DIM;
  for (int i = 0; i < 3; ++i) {
    int j = tid + 256 * i;
    const bf16* wr = We2 + (long long)j * (2 * DIM);
    float acc = be2[j];
    for (int k = 0; k < 2 * DIM; ++k) acc += s_h[k] * (float)wr[k];
    out[j] = (bf16)(s_p[j] + acc);
  }
}

__global__ void fill_centers(const float* __restrict__ pc, bf16* __restrict__ combined) {
  int i = blockIdx.x * 256 + threadIdx.x;     // over 16*32*768
  if (i >= BQ * 32 * DIM) return;
  int j = i % DIM;
  int t = (i / DIM) % 32;
  int b = i / (DIM * 32);
  combined[((long long)b * TT + 32 + t) * DIM + j] = (bf16)pc[t * DIM + j];
}

// ---------------------------------------------------------------------------
// KV: kvrow = combined_row @ Wkv^T; K l2-normalized * learn_scale[h];
// K stored [B,H,T,hd], V transposed [B,H,hd,T] (WMMA B-fragment friendly).
// ---------------------------------------------------------------------------
__global__ __launch_bounds__(256) void kv_kernel(
    const bf16* __restrict__ combined, const bf16* __restrict__ Wkv,
    const float* __restrict__ ls, bf16* __restrict__ kn, bf16* __restrict__ vT) {
  __shared__ float s_c[DIM];
  __shared__ float s_kv[2 * DIM];
  __shared__ float s_n[NH];
  const int tid = threadIdx.x;
  const int row = blockIdx.x;            // b*64 + t
  const int b = row >> 6, t = row & 63;
  const bf16* cr = combined + (long long)row * DIM;
  for (int i = tid; i < DIM; i += 256) s_c[i] = (float)cr[i];
  __syncthreads();
  for (int i = 0; i < 6; ++i) {
    int j = tid + 256 * i;
    const bf16* wr = Wkv + (long long)j * DIM;
    float acc = 0.0f;
    for (int k = 0; k < DIM; ++k) acc += s_c[k] * (float)wr[k];
    s_kv[j] = acc;
  }
  __syncthreads();
  if (tid < NH) {
    float ss = 0.0f;
    for (int d = 0; d < HD; ++d) { float v = s_kv[tid * HD + d]; ss += v * v; }
    s_n[tid] = ls[tid] / fmaxf(sqrtf(ss), 1e-12f);
  }
  __syncthreads();
  for (int e = tid; e < DIM; e += 256) {
    int h = e >> 6, d = e & 63;
    long long base = (long long)(b * NH + h) * (TT * HD);
    kn[base + (long long)t * HD + d] = (bf16)(s_kv[e] * s_n[h]);
    vT[base + (long long)d * TT + t] = (bf16)s_kv[DIM + e];
  }
}

// ---------------------------------------------------------------------------
// Attention: per wave one (b,h,32-row tile): attn = silu(q@kn^T); out=attn@v.
// attn staged through LDS (padded) to re-layout into A-fragments.
// ---------------------------------------------------------------------------
#define APAD 72
__global__ __launch_bounds__(256) void attn_kernel(
    const bf16* __restrict__ qn, const bf16* __restrict__ kn,
    const bf16* __restrict__ vT, bf16* __restrict__ ao) {
  __shared__ bf16 s_attn[8][32 * APAD];
  const int tid = threadIdx.x, lane = tid & 31, wv = tid >> 5;
  const int kg = lane >> 4, ln16 = lane & 15;
  const int wid = blockIdx.x * 8 + wv;       // 16*12*128 wave-tiles
  const int ntile = wid & 127;
  const int bh = wid >> 7;
  const int h = bh % NH;
  const int b = bh / NH;
  const long long qrow0 = (long long)b * NSEQ + ntile * 32;
  const bf16* Q  = qn + qrow0 * DIM + h * HD;
  const bf16* Kh = kn + (long long)bh * (TT * HD);   // [t][d]
  const bf16* Vh = vT + (long long)bh * (TT * HD);   // [d][t]

  v8f acc[2][4] = {};
#pragma unroll
  for (int k0 = 0; k0 < HD; k0 += 32) {
    v16bf a0 = load_a_frag(Q + k0, DIM);
    v16bf a1 = load_a_frag(Q + 16LL * DIM + k0, DIM);
#pragma unroll
    for (int j = 0; j < 4; ++j) {
      v16bf bfrag = load_b_frag(Kh + (long long)(j * 16) * HD + k0, HD);
      acc[0][j] = wmma_bf16(a0, bfrag, acc[0][j]);
      acc[1][j] = wmma_bf16(a1, bfrag, acc[1][j]);
    }
  }
  bf16* sa = &s_attn[wv][0];
#pragma unroll
  for (int mt = 0; mt < 2; ++mt)
#pragma unroll
    for (int r = 0; r < 8; ++r)
#pragma unroll
      for (int nt = 0; nt < 4; ++nt)
        sa[(mt * 16 + r + 8 * kg) * APAD + nt * 16 + ln16] =
            (bf16)silu_f(acc[mt][nt][r]);
  __syncthreads();

  v8f oacc[2][4] = {};
#pragma unroll
  for (int k0 = 0; k0 < TT; k0 += 32) {
    v16bf a0 = load_a_frag(sa + k0, APAD);
    v16bf a1 = load_a_frag(sa + 16 * APAD + k0, APAD);
#pragma unroll
    for (int j = 0; j < 4; ++j) {
      v16bf bfrag = load_b_frag(Vh + (long long)(j * 16) * TT + k0, TT);
      oacc[0][j] = wmma_bf16(a0, bfrag, oacc[0][j]);
      oacc[1][j] = wmma_bf16(a1, bfrag, oacc[1][j]);
    }
  }
  bf16* O = ao + qrow0 * DIM + h * HD;
#pragma unroll
  for (int mt = 0; mt < 2; ++mt)
#pragma unroll
    for (int r = 0; r < 8; ++r)
#pragma unroll
      for (int nt = 0; nt < 4; ++nt)
        O[(long long)(mt * 16 + r + 8 * kg) * DIM + nt * 16 + ln16] =
            (bf16)oacc[mt][nt][r];
}

// ---------------------------------------------------------------------------
// Generic WMMA GEMM: C[M,Nc] = A[M,K] @ W[Nc,K]^T (+bias)(+epilogue).
// Double-buffered LDS staging (async global->LDS when available).
// Block tile 128x128 (8 waves 4Mx2N), wave tile 32x64, K step 32.
// MODE 0: bf16 out (+bias); MODE 1: bf16 out + bias + GELU;
// MODE 2: f32 out + bias + bf16 residual; MODE 3: bf16 out, per-64-col l2norm.
// ---------------------------------------------------------------------------
template <int MODE>
__global__ __launch_bounds__(256) void gemm_wmma(
    const bf16* __restrict__ A, const bf16* __restrict__ W,
    const float* __restrict__ bias, const bf16* __restrict__ resid,
    bf16* __restrict__ outB, float* __restrict__ outF, int Nc, int K) {
  __shared__ bf16 sA[2][128 * LDP];
  __shared__ bf16 sB[2][128 * LDP];
  const int tid = threadIdx.x, lane = tid & 31, wv = tid >> 5;
  const int wm = wv >> 1, wn = wv & 1;
  const int kg = lane >> 4, ln16 = lane & 15;
  const long long row0g = (long long)blockIdx.x * 128;
  const int col0g = blockIdx.y * 128;
  const int wrow = wm * 32;      // wave row offset in block
  const int wcol = wn * 64;      // wave col offset in block

  // Stage one 128x32 slice of A and of W into LDS buffer `buf`.
  // 512 16B-chunks per matrix; 256 threads x 2 chunks; 4 copies/thread/stage.
  auto stage = [&](int buf, int k0) {
#pragma unroll
    for (int i = 0; i < 2; ++i) {
      int c = tid + 256 * i;         // 0..511
      int r = c >> 2, kc = c & 3;    // row in tile, 16B chunk in k-slice
      copy16_g2l(A + (row0g + r) * (long long)K + k0 + kc * 8,
                 &sA[buf][r * LDP + kc * 8]);
      copy16_g2l(W + (long long)(col0g + r) * K + k0 + kc * 8,
                 &sB[buf][r * LDP + kc * 8]);
    }
  };

  v8f acc[2][4] = {};
  const int nk = K / 32;
  stage(0, 0);
  stage(1, 32);

  for (int ks = 0; ks < nk; ++ks) {
    const int buf = ks & 1;
    if (ks + 1 < nk) wait_async<4>(); else wait_async<0>();
    __syncthreads();   // all waves' staged data visible

    const bf16* aB = &sA[buf][wrow * LDP];
    v16bf a0 = load_a_frag(aB, LDP);
    v16bf a1 = load_a_frag(aB + 16 * LDP, LDP);
#pragma unroll
    for (int j = 0; j < 4; ++j) {
      v16bf bfrag = load_b_frag(&sB[buf][(wcol + j * 16) * LDP], LDP);
      acc[0][j] = wmma_bf16(a0, bfrag, acc[0][j]);
      acc[1][j] = wmma_bf16(a1, bfrag, acc[1][j]);
    }
    __syncthreads();   // done reading `buf` before it is refilled
    if (ks + 2 < nk) stage(buf, (ks + 2) * 32);
  }

  if constexpr (MODE == 3) {
    // per-head (64-col) L2 normalization; wave tile spans exactly one head.
#pragma unroll
    for (int mt = 0; mt < 2; ++mt) {
#pragma unroll
      for (int r = 0; r < 8; ++r) {
        float s = 0.0f;
#pragma unroll
        for (int nt = 0; nt < 4; ++nt) { float v = acc[mt][nt][r]; s += v * v; }
        s += __shfl_xor(s, 1); s += __shfl_xor(s, 2);
        s += __shfl_xor(s, 4); s += __shfl_xor(s, 8);
        float rn = 1.0f / fmaxf(sqrtf(s), 1e-12f);
        long long m = row0g + wrow + mt * 16 + r + 8 * kg;
#pragma unroll
        for (int nt = 0; nt < 4; ++nt)
          outB[m * Nc + col0g + wcol + nt * 16 + ln16] =
              (bf16)(acc[mt][nt][r] * rn);
      }
    }
  } else {
#pragma unroll
    for (int mt = 0; mt < 2; ++mt) {
#pragma unroll
      for (int r = 0; r < 8; ++r) {
        long long m = row0g + wrow + mt * 16 + r + 8 * kg;
#pragma unroll
        for (int nt = 0; nt < 4; ++nt) {
          int c = col0g + wcol + nt * 16 + ln16;
          float v = acc[mt][nt][r];
          if (bias) v += bias[c];
          if constexpr (MODE == 1) v = gelu_f(v);
          if constexpr (MODE == 2) {
            outF[m * Nc + c] = v + (float)resid[m * Nc + c];
          } else {
            outB[m * Nc + c] = (bf16)v;
          }
        }
      }
    }
  }
}

// ---------------------------------------------------------------------------
// Host driver
// ---------------------------------------------------------------------------
extern "C" void kernel_launch(void* const* d_in, const int* in_sizes, int n_in,
                              void* d_out, int out_size, void* d_ws, size_t ws_size,
                              hipStream_t stream) {
  (void)in_sizes; (void)n_in; (void)out_size; (void)ws_size;
  const float* x      = (const float*)d_in[0];
  const float* proto  = (const float*)d_in[1];
  const float* n1w    = (const float*)d_in[2];
  const float* n1b    = (const float*)d_in[3];
  const float* n2w    = (const float*)d_in[4];
  const float* n2b    = (const float*)d_in[5];
  const float* Wq     = (const float*)d_in[6];
  const float* Wkv    = (const float*)d_in[7];
  const float* lscale = (const float*)d_in[8];
  const float* pcent  = (const float*)d_in[9];
  const float* We1    = (const float*)d_in[10];
  const float* be1    = (const float*)d_in[11];
  const float* We2    = (const float*)d_in[12];
  const float* be2    = (const float*)d_in[13];
  const float* Wproj  = (const float*)d_in[14];
  const float* bproj  = (const float*)d_in[15];
  const float* Wfc1   = (const float*)d_in[16];
  const float* bfc1   = (const float*)d_in[17];
  const float* Wfc2   = (const float*)d_in[18];
  const float* bfc2   = (const float*)d_in[19];
  float* out = (float*)d_out;

  // ---- workspace layout (~630 MB, 256B aligned) ----
  char* ws = (char*)d_ws;
  size_t off = 0;
  auto take = [&](size_t bytes) -> char* {
    off = (off + 255) & ~(size_t)255;
    char* p = ws + off;
    off += bytes;
    return p;
  };
  bf16* wq_b   = (bf16*)take((size_t)DIM * DIM * 2);
  bf16* wkv_b  = (bf16*)take((size_t)2 * DIM * DIM * 2);
  bf16* we1_b  = (bf16*)take((size_t)2 * DIM * DIM * 2);
  bf16* we2_b  = (bf16*)take((size_t)2 * DIM * DIM * 2);
  bf16* wpr_b  = (bf16*)take((size_t)DIM * DIM * 2);
  bf16* wf1_b  = (bf16*)take((size_t)4 * DIM * DIM * 2);
  bf16* wf2_b  = (bf16*)take((size_t)4 * DIM * DIM * 2);
  bf16* pn     = (bf16*)take((size_t)BQ * 32 * DIM * 2);
  bf16* comb   = (bf16*)take((size_t)BQ * TT * DIM * 2);
  bf16* kn     = (bf16*)take((size_t)BQ * NH * TT * HD * 2);
  bf16* vT     = (bf16*)take((size_t)BQ * NH * TT * HD * 2);
  bf16* ybf    = (bf16*)take((size_t)MROWS * DIM * 2);       // live to the end
  bf16* xn     = (bf16*)take((size_t)MROWS * DIM * 2);       // dead after Q
  bf16* big    = (bf16*)take((size_t)MROWS * 4 * DIM * 2);   // 384 MB region
  bf16* qn   = big;                                  // dead after attention
  bf16* ao   = big + (size_t)MROWS * DIM;            // dead after proj
  bf16* z    = xn;                                   // LN2 output (xn dead)
  bf16* mlp1 = big;                                  // [M,3072] (qn/ao dead)

  // ---- 1. weights -> bf16 ----
  auto cvt = [&](const float* s, bf16* d, int n) {
    cvt_f32_bf16<<<(n + 255) / 256, 256, 0, stream>>>(s, d, n);
  };
  cvt(Wq,    wq_b,  DIM * DIM);
  cvt(Wkv,   wkv_b, 2 * DIM * DIM);
  cvt(We1,   we1_b, 2 * DIM * DIM);
  cvt(We2,   we2_b, 2 * DIM * DIM);
  cvt(Wproj, wpr_b, DIM * DIM);
  cvt(Wfc1,  wf1_b, 4 * DIM * DIM);
  cvt(Wfc2,  wf2_b, 4 * DIM * DIM);

  // ---- 2. LayerNorm ----
  ln_kernel<float><<<MROWS, 256, 0, stream>>>(x, n1w, n1b, xn);
  ln_kernel<float><<<BQ * 32, 256, 0, stream>>>(proto, n1w, n1b, pn);

  // ---- 3. prototype enhancer + centers ----
  proto_enhance<<<BQ * 32, 256, 0, stream>>>(pn, we1_b, be1, we2_b, be2, comb);
  fill_centers<<<(BQ * 32 * DIM + 255) / 256, 256, 0, stream>>>(pcent, comb);

  // ---- 4. K/V ----
  kv_kernel<<<BQ * TT, 256, 0, stream>>>(comb, wkv_b, lscale, kn, vT);

  // ---- 5. Q = xn @ Wq^T, per-head l2norm (MODE 3) ----
  gemm_wmma<3><<<dim3(MROWS / 128, DIM / 128), 256, 0, stream>>>(
      xn, wq_b, nullptr, nullptr, qn, nullptr, DIM, DIM);

  // ---- 6. attention ----
  attn_kernel<<<BQ * NH * (NSEQ / 32) / 8, 256, 0, stream>>>(qn, kn, vT, ao);

  // ---- 7. proj ----
  gemm_wmma<0><<<dim3(MROWS / 128, DIM / 128), 256, 0, stream>>>(
      ao, wpr_b, bproj, nullptr, ybf, nullptr, DIM, DIM);

  // ---- 8. LN2 ----
  ln_kernel<bf16><<<MROWS, 256, 0, stream>>>(ybf, n2w, n2b, z);

  // ---- 9. fc1 (GELU) ----
  gemm_wmma<1><<<dim3(MROWS / 128, 4 * DIM / 128), 256, 0, stream>>>(
      z, wf1_b, bfc1, nullptr, mlp1, nullptr, 4 * DIM, DIM);

  // ---- 10. fc2 + residual(y) -> f32 out ----
  gemm_wmma<2><<<dim3(MROWS / 128, DIM / 128), 256, 0, stream>>>(
      mlp1, wf2_b, bfc2, ybf, nullptr, out, DIM, 4 * DIM);
}
